// SymQuantMatMul_42520176230457
// MI455X (gfx1250) — 8-bit matrix operand order repaired and checked by reading, compiled, not run on hardware
//
#include <hip/hip_runtime.h>

typedef __attribute__((ext_vector_type(8))) int v8i;

#define M_DIM 8192
#define N_DIM 4096
#define K_DIM 4096
#define QMAXF 127.0f

#define MB 256
#define NB 128
#define KC 64
#define LDS_STRIDE 80
#define BUF_BYTES ((MB + NB) * LDS_STRIDE)

#if defined(__has_builtin)
#if __has_builtin(__builtin_amdgcn_global_load_async_to_lds_b128)
#define USE_ASYNC_LDS 1
#endif
#endif
#ifndef USE_ASYNC_LDS
#define USE_ASYNC_LDS 0
#endif

#if USE_ASYNC_LDS
typedef int v4i_t __attribute__((vector_size(16)));
typedef __attribute__((address_space(1))) v4i_t* as1_v4i_p;
typedef __attribute__((address_space(3))) v4i_t* as3_v4i_p;
#define ASYNC_B128(gptr, lptr)                    \
  __builtin_amdgcn_global_load_async_to_lds_b128( \
      (as1_v4i_p)(gptr), (as3_v4i_p)(lptr), 0, 0)
#if __has_builtin(__builtin_amdgcn_s_wait_asynccnt)
#define WAIT_ASYNC() __builtin_amdgcn_s_wait_asynccnt(0)
#else
#define WAIT_ASYNC() asm volatile("s_wait_asynccnt 0x0" ::: "memory")
#endif
#endif

__global__ __launch_bounds__(256) void quant_rows_kernel(
    const float* __restrict__ X, signed char* __restrict__ Q,
    float* __restrict__ S) {
  __shared__ float red[8];
  const int row = blockIdx.x;
  const int t = threadIdx.x;
  const float* xrow = X + (size_t)row * K_DIM;
  const float4* xrow4 = (const float4*)xrow;

  float4 vals[4];
#pragma unroll
  for (int i = 0; i < 4; ++i) vals[i] = xrow4[t + 256 * i];

  float mx = 0.0f;
#pragma unroll
  for (int i = 0; i < 4; ++i) {
    mx = fmaxf(mx, fabsf(vals[i].x));
    mx = fmaxf(mx, fabsf(vals[i].y));
    mx = fmaxf(mx, fabsf(vals[i].z));
    mx = fmaxf(mx, fabsf(vals[i].w));
  }
#pragma unroll
  for (int off = 16; off > 0; off >>= 1)
    mx = fmaxf(mx, __shfl_xor(mx, off, 32));
  if ((t & 31) == 0) red[t >> 5] = mx;
  __syncthreads();
  float m = fmaxf(fmaxf(fmaxf(red[0], red[1]), fmaxf(red[2], red[3])),
                  fmaxf(fmaxf(red[4], red[5]), fmaxf(red[6], red[7])));
  const float scale = fmaxf(m / QMAXF, 1e-8f);
  const float rs = 1.0f / scale;
  if (t == 0) S[row] = scale;

  int* qrow = (int*)(Q + (size_t)row * K_DIM);
#pragma unroll
  for (int i = 0; i < 4; ++i) {
    const float4 v = vals[i];
    int q0 = (int)rintf(v.x * rs);
    int q1 = (int)rintf(v.y * rs);
    int q2 = (int)rintf(v.z * rs);
    int q3 = (int)rintf(v.w * rs);
    q0 = q0 < -128 ? -128 : (q0 > 127 ? 127 : q0);
    q1 = q1 < -128 ? -128 : (q1 > 127 ? 127 : q1);
    q2 = q2 < -128 ? -128 : (q2 > 127 ? 127 : q2);
    q3 = q3 < -128 ? -128 : (q3 > 127 ? 127 : q3);
    qrow[t + 256 * i] = (q0 & 0xFF) | ((q1 & 0xFF) << 8) | ((q2 & 0xFF) << 16) |
                        ((q3 & 0xFF) << 24);
  }
}

__global__ __launch_bounds__(256) void int8_wmma_gemm_kernel(
    const signed char* __restrict__ qA, const signed char* __restrict__ qB,
    const float* __restrict__ sA, const float* __restrict__ sB,
    float* __restrict__ out) {
  __shared__ signed char lds[2 * BUF_BYTES];

  const int t = threadIdx.x;
  const int lane = t & 31;
  const int w = t >> 5;
  const int l = lane & 15;
  const int h = lane >> 4;
  const int wm = (w & 3) * 64;
  const int wn = (w >> 2) * 64;
  const int bm = blockIdx.y * MB;
  const int bn = blockIdx.x * NB;

  const int srow = t >> 2;
  const int scol = (t & 3) * 16;
  const signed char* gA = qA + (size_t)(bm + srow) * K_DIM + scol;
  const signed char* gB = qB + (size_t)(bn + srow) * K_DIM + scol;

#if USE_ASYNC_LDS
#pragma unroll
  for (int i = 0; i < 4; ++i)
    ASYNC_B128(gA + (size_t)(64 * i) * K_DIM,
               lds + (srow + 64 * i) * LDS_STRIDE + scol);
#pragma unroll
  for (int i = 0; i < 2; ++i)
    ASYNC_B128(gB + (size_t)(64 * i) * K_DIM,
               lds + MB * LDS_STRIDE + (srow + 64 * i) * LDS_STRIDE + scol);
  WAIT_ASYNC();
#else
#pragma unroll
  for (int i = 0; i < 4; ++i) {
    int4 v = *(const int4*)(gA + (size_t)(64 * i) * K_DIM);
    *(int4*)(lds + (srow + 64 * i) * LDS_STRIDE + scol) = v;
  }
#pragma unroll
  for (int i = 0; i < 2; ++i) {
    int4 v = *(const int4*)(gB + (size_t)(64 * i) * K_DIM);
    *(int4*)(lds + MB * LDS_STRIDE + (srow + 64 * i) * LDS_STRIDE + scol) = v;
  }
#endif
  __syncthreads();

  v8i acc[4][4];
#pragma unroll
  for (int i = 0; i < 4; ++i)
#pragma unroll
    for (int j = 0; j < 4; ++j) acc[i][j] = {};

  const int NK = K_DIM / KC;
  for (int s = 0; s < NK; ++s) {
    const signed char* buf = lds + (s & 1) * BUF_BYTES;
    const signed char* bufB = buf + MB * LDS_STRIDE;
    const bool more = (s + 1) < NK;

#if USE_ASYNC_LDS
    if (more) {
      signed char* nbuf = lds + ((s + 1) & 1) * BUF_BYTES;
      const int kk = (s + 1) * KC;
#pragma unroll
      for (int i = 0; i < 4; ++i)
        ASYNC_B128(gA + (size_t)(64 * i) * K_DIM + kk,
                   nbuf + (srow + 64 * i) * LDS_STRIDE + scol);
#pragma unroll
      for (int i = 0; i < 2; ++i)
        ASYNC_B128(gB + (size_t)(64 * i) * K_DIM + kk,
                   nbuf + MB * LDS_STRIDE + (srow + 64 * i) * LDS_STRIDE + scol);
    }
#else
    int4 pa[4], pb[2];
    if (more) {
      const int kk = (s + 1) * KC;
#pragma unroll
      for (int i = 0; i < 4; ++i)
        pa[i] = *(const int4*)(gA + (size_t)(64 * i) * K_DIM + kk);
#pragma unroll
      for (int i = 0; i < 2; ++i)
        pb[i] = *(const int4*)(gB + (size_t)(64 * i) * K_DIM + kk);
    }
#endif

    v8i bf[4];
#pragma unroll
    for (int sn = 0; sn < 4; ++sn) {
      const signed char* p = bufB + (wn + sn * 16 + l) * LDS_STRIDE + 16 * h;
      int4 t0 = *(const int4*)(p);
      int4 t1 = *(const int4*)(p + 32);
      bf[sn][0] = t0.x; bf[sn][1] = t0.y; bf[sn][2] = t0.z; bf[sn][3] = t0.w;
      bf[sn][4] = t1.x; bf[sn][5] = t1.y; bf[sn][6] = t1.z; bf[sn][7] = t1.w;
    }

#pragma unroll
    for (int sm = 0; sm < 4; ++sm) {
      const signed char* p = buf + (wm + sm * 16 + l) * LDS_STRIDE + 16 * h;
      v8i af;
#pragma unroll
      for (int q = 0; q < 4; ++q) {
        int2 tv = *(const int2*)(p + 8 * q + 16 * (q >> 1));
        af[2 * q] = tv.x;
        af[2 * q + 1] = tv.y;
      }
#pragma unroll
      for (int sn = 0; sn < 4; ++sn)
        acc[sm][sn] = __builtin_amdgcn_wmma_i32_16x16x64_iu8(
            true, af, true, bf[sn], acc[sm][sn], false, false);
    }

    if (more) {
#if USE_ASYNC_LDS
      WAIT_ASYNC();
#else
      signed char* nbuf = lds + ((s + 1) & 1) * BUF_BYTES;
#pragma unroll
      for (int i = 0; i < 4; ++i)
        *(int4*)(nbuf + (srow + 64 * i) * LDS_STRIDE + scol) = pa[i];
#pragma unroll
      for (int i = 0; i < 2; ++i)
        *(int4*)(nbuf + MB * LDS_STRIDE + (srow + 64 * i) * LDS_STRIDE + scol) =
            pb[i];
#endif
      __syncthreads();
    }
  }

#pragma unroll
  for (int sm = 0; sm < 4; ++sm) {
    const int mbase = bm + wm + sm * 16;
    float sa[8];
#pragma unroll
    for (int r = 0; r < 8; ++r) sa[r] = sA[mbase + r + 8 * h];
#pragma unroll
    for (int sn = 0; sn < 4; ++sn) {
      const int col = bn + wn + sn * 16 + l;
      const float sb = sB[col];
#pragma unroll
      for (int r = 0; r < 8; ++r) {
        const int row = mbase + r + 8 * h;
        out[(size_t)row * N_DIM + col] = (float)acc[sm][sn][r] * sa[r] * sb;
      }
    }
  }
}

extern "C" void kernel_launch(void* const* d_in, const int* in_sizes, int n_in,
                              void* d_out, int out_size, void* d_ws,
                              size_t ws_size, hipStream_t stream) {
  (void)in_sizes; (void)n_in; (void)out_size; (void)ws_size;
  const float* A = (const float*)d_in[0];
  const float* B = (const float*)d_in[1];
  float* out = (float*)d_out;

  signed char* qA = (signed char*)d_ws;
  signed char* qB = qA + (size_t)M_DIM * K_DIM;
  float* sA = (float*)(qB + (size_t)N_DIM * K_DIM);
  float* sB = sA + M_DIM;

  quant_rows_kernel<<<M_DIM, 256, 0, stream>>>(A, qA, sA);
  quant_rows_kernel<<<N_DIM, 256, 0, stream>>>(B, qB, sB);

  dim3 grid(N_DIM / NB, M_DIM / MB);
  int8_wmma_gemm_kernel<<<grid, 256, 0, stream>>>(qA, qB, sA, sB, out);
}
